// Transcoder_63891933495741
// MI455X (gfx1250) — matrix operand order repaired and checked by reading, not compiled, not run on hardware
//
#include <hip/hip_runtime.h>
#include <hip/hip_bf16.h>

#define N_TOKENS 4096
#define D_MODEL  1024
#define D_HIDDEN 16384
#define TOPK     64

#define BM 128
#define BN 128
#define BK 32
#define LDSTR 40   // ushorts per LDS row: 32 data + 8 pad (80B, 16B-aligned)

typedef __bf16 v16bf __attribute__((ext_vector_type(16)));
typedef float  v8f   __attribute__((ext_vector_type(8)));

union Frag { v16bf v; uint4 q[2]; };
union Acc  { v8f v; float f[8]; };

__device__ __forceinline__ unsigned short f2bf_rne(float f) {
    unsigned u = __float_as_uint(f);
    unsigned r = u + 0x7FFFu + ((u >> 16) & 1u);
    return (unsigned short)(r >> 16);
}

__device__ __forceinline__ uint4 pack8(const unsigned short* s) {
    uint4 q;
    q.x = (unsigned)s[0] | ((unsigned)s[1] << 16);
    q.y = (unsigned)s[2] | ((unsigned)s[3] << 16);
    q.z = (unsigned)s[4] | ((unsigned)s[5] << 16);
    q.w = (unsigned)s[6] | ((unsigned)s[7] << 16);
    return q;
}

// ---------------- Kernel A: z = relu(x @ W_enc + b_enc) via split-bf16 WMMA ----
__global__ __launch_bounds__(256)
void encode_gemm_relu(const float* __restrict__ X, const float* __restrict__ W,
                      const float* __restrict__ bias, float* __restrict__ Z) {
    __shared__ unsigned short As_hi[BM * LDSTR];
    __shared__ unsigned short As_lo[BM * LDSTR];
    __shared__ unsigned short Bs_hi[BN * LDSTR];
    __shared__ unsigned short Bs_lo[BN * LDSTR];
    __shared__ float bias_s[BN];

    const int t    = threadIdx.x;
    const int lane = t & 31;
    const int w    = t >> 5;
    const int lm   = lane & 15;
    const int g    = lane >> 4;            // lane group 0/1
    const int wm   = (w >> 2) * 64;        // wave m-offset: 0 or 64
    const int wn   = (w & 3) * 32;         // wave n-offset: 0..96
    const int bn0  = blockIdx.x * BN;
    const int bm0  = blockIdx.y * BM;

    if (t < BN) bias_s[t] = bias[bn0 + t];

    Acc acc[4][2];
    #pragma unroll
    for (int i = 0; i < 4; ++i)
        #pragma unroll
        for (int j = 0; j < 2; ++j)
            #pragma unroll
            for (int r = 0; r < 8; ++r) acc[i][j].f[r] = 0.0f;

    const int bl_n  = t & 127;   // B-load: column within tile
    const int bl_kh = t >> 7;    // B-load: k half (0/1)

    for (int k0 = 0; k0 < D_MODEL; k0 += BK) {
        __syncthreads();
        // ---- A tile 128x32 f32 -> LDS hi/lo bf16 (row-major [m][k]) ----
        #pragma unroll
        for (int it = 0; it < 4; ++it) {
            int idx = t + 256 * it;              // float4 slot 0..1023
            int row = idx >> 3;
            int c4  = idx & 7;
            const float4 v = *(const float4*)(X + (size_t)(bm0 + row) * D_MODEL + k0 + c4 * 4);
            unsigned u0 = __float_as_uint(v.x), u1 = __float_as_uint(v.y);
            unsigned u2 = __float_as_uint(v.z), u3 = __float_as_uint(v.w);
            uint2 hi, lo;
            hi.x = (u0 >> 16) | (u1 & 0xFFFF0000u);
            hi.y = (u2 >> 16) | (u3 & 0xFFFF0000u);
            lo.x = (unsigned)f2bf_rne(v.x - __uint_as_float(u0 & 0xFFFF0000u)) |
                   ((unsigned)f2bf_rne(v.y - __uint_as_float(u1 & 0xFFFF0000u)) << 16);
            lo.y = (unsigned)f2bf_rne(v.z - __uint_as_float(u2 & 0xFFFF0000u)) |
                   ((unsigned)f2bf_rne(v.w - __uint_as_float(u3 & 0xFFFF0000u)) << 16);
            *(uint2*)&As_hi[row * LDSTR + c4 * 4] = hi;
            *(uint2*)&As_lo[row * LDSTR + c4 * 4] = lo;
        }
        // ---- B tile 32x128 f32 -> LDS hi/lo bf16 transposed [n][k] ----
        {
            unsigned short bh[16], bl[16];
            #pragma unroll
            for (int kk = 0; kk < 16; ++kk) {
                int k = bl_kh * 16 + kk;
                float v = W[(size_t)(k0 + k) * D_HIDDEN + bn0 + bl_n];
                unsigned u = __float_as_uint(v);
                bh[kk] = (unsigned short)(u >> 16);
                bl[kk] = f2bf_rne(v - __uint_as_float(u & 0xFFFF0000u));
            }
            unsigned short* ph = &Bs_hi[bl_n * LDSTR + bl_kh * 16];
            unsigned short* pl = &Bs_lo[bl_n * LDSTR + bl_kh * 16];
            *(uint4*)(ph)     = pack8(bh);
            *(uint4*)(ph + 8) = pack8(bh + 8);
            *(uint4*)(pl)     = pack8(bl);
            *(uint4*)(pl + 8) = pack8(bl + 8);
        }
        __syncthreads();

        // ---- fragments ----
        Frag ah[4], al[4], bhf[2], blf[2];
        #pragma unroll
        for (int i = 0; i < 4; ++i) {
            int m = wm + i * 16 + lm;
            const unsigned short* ph = &As_hi[m * LDSTR];
            const unsigned short* pl = &As_lo[m * LDSTR];
            ah[i].q[0] = *(const uint4*)(ph + g * 8);
            ah[i].q[1] = *(const uint4*)(ph + 16 + g * 8);
            al[i].q[0] = *(const uint4*)(pl + g * 8);
            al[i].q[1] = *(const uint4*)(pl + 16 + g * 8);
        }
        #pragma unroll
        for (int j = 0; j < 2; ++j) {
            int n = wn + j * 16 + lm;
            const unsigned short* ph = &Bs_hi[n * LDSTR + g * 8];
            const unsigned short* pl = &Bs_lo[n * LDSTR + g * 8];
            bhf[j].q[0] = *(const uint4*)(ph);
            bhf[j].q[1] = *(const uint4*)(ph + 16);
            blf[j].q[0] = *(const uint4*)(pl);
            blf[j].q[1] = *(const uint4*)(pl + 16);
        }
        // ---- split-precision MMA: hi*hi + hi*lo + lo*hi ----
        #pragma unroll
        for (int i = 0; i < 4; ++i)
            #pragma unroll
            for (int j = 0; j < 2; ++j) {
                acc[i][j].v = __builtin_amdgcn_wmma_f32_16x16x32_bf16(
                    false, ah[i].v, false, bhf[j].v, (short)0, acc[i][j].v, false, false);
                acc[i][j].v = __builtin_amdgcn_wmma_f32_16x16x32_bf16(
                    false, ah[i].v, false, blf[j].v, (short)0, acc[i][j].v, false, false);
                acc[i][j].v = __builtin_amdgcn_wmma_f32_16x16x32_bf16(
                    false, al[i].v, false, bhf[j].v, (short)0, acc[i][j].v, false, false);
            }
    }

    // ---- epilogue: bias + relu, C/D layout: lane->(N=lm, M=g*8+r) ----
    #pragma unroll
    for (int i = 0; i < 4; ++i)
        #pragma unroll
        for (int j = 0; j < 2; ++j) {
            int nloc = wn + j * 16 + lm;
            int n = bn0 + nloc;
            float b = bias_s[nloc];
            #pragma unroll
            for (int r = 0; r < 8; ++r) {
                int m = bm0 + wm + i * 16 + g * 8 + r;
                float v = acc[i][j].f[r] + b;
                Z[(size_t)m * D_HIDDEN + n] = v > 0.0f ? v : 0.0f;
            }
        }
}

// ---------------- Kernel B: per-row top-64 via radix select on f32 bits ------
__global__ __launch_bounds__(256)
void topk_select(const float* __restrict__ Z, float* __restrict__ tv, int* __restrict__ ti) {
    const int row = blockIdx.x;
    const int t   = threadIdx.x;
    const float* zr = Z + (size_t)row * D_HIDDEN;
    __shared__ unsigned hist[256];
    __shared__ int s_bin, s_need, s_cnt, s_cnt2;

    unsigned prefix = 0;
    int need = TOPK;
    for (int level = 0; level < 4; ++level) {
        hist[t] = 0u;
        __syncthreads();
        const int shift = 24 - 8 * level;
        for (int i = t; i < D_HIDDEN; i += 256) {
            unsigned u = __float_as_uint(zr[i]);
            bool ok = (level == 0) ? true : ((u >> (shift + 8)) == prefix);
            if (ok) atomicAdd(&hist[(u >> shift) & 255u], 1u);
        }
        __syncthreads();
        if (t == 0) {
            int b = 256; unsigned cum = 0;
            while (b > 0) { --b; cum += hist[b]; if ((int)cum >= need) break; }
            s_bin  = b;
            s_need = need - (int)(cum - hist[b]);
        }
        __syncthreads();
        prefix = (prefix << 8) | (unsigned)s_bin;
        need   = s_need;
    }
    const unsigned T = prefix;

    // pre-zero outputs so short rows contribute nothing in decode
    if (t < TOPK) { tv[row * TOPK + t] = 0.0f; ti[row * TOPK + t] = 0; }
    if (t == 0) s_cnt = 0;
    __syncthreads();
    for (int i = t; i < D_HIDDEN; i += 256) {
        float z = zr[i];
        unsigned u = __float_as_uint(z);
        if (u > T) {
            int p = atomicAdd(&s_cnt, 1);
            tv[row * TOPK + p] = z;
            ti[row * TOPK + p] = i;
        }
    }
    __syncthreads();
    const int base = s_cnt;
    if (t == 0) s_cnt2 = 0;
    __syncthreads();
    for (int i = t; i < D_HIDDEN; i += 256) {
        float z = zr[i];
        unsigned u = __float_as_uint(z);
        if (u == T) {
            int p = atomicAdd(&s_cnt2, 1);
            if (base + p < TOPK) {
                tv[row * TOPK + base + p] = z;
                ti[row * TOPK + base + p] = i;
            }
        }
    }
}

// ---------------- Kernel C: y = z_sparse @ W_dec + b_dec ---------------------
__global__ __launch_bounds__(256)
void decode_sparse(const float* __restrict__ tv, const int* __restrict__ ti,
                   const float* __restrict__ Wd, const float* __restrict__ bd,
                   float* __restrict__ Y) {
    const int row = blockIdx.x;
    const int t   = threadIdx.x;
    __shared__ float sv[TOPK];
    __shared__ int   si[TOPK];
    if (t < TOPK) { sv[t] = tv[row * TOPK + t]; si[t] = ti[row * TOPK + t]; }
    __syncthreads();

    const int c = t * 4;                       // 256 threads * 4 = 1024 cols
    float4 accv = *(const float4*)(bd + c);
    #pragma unroll 4
    for (int j = 0; j < TOPK; ++j) {
        if (j + 1 < TOPK)
            __builtin_prefetch(Wd + (size_t)si[j + 1] * D_MODEL + c, 0, 0);
        const float s = sv[j];
        const float4 wv = *(const float4*)(Wd + (size_t)si[j] * D_MODEL + c);
        accv.x += s * wv.x; accv.y += s * wv.y;
        accv.z += s * wv.z; accv.w += s * wv.w;
    }
    *(float4*)(Y + (size_t)row * D_MODEL + c) = accv;
}

extern "C" void kernel_launch(void* const* d_in, const int* in_sizes, int n_in,
                              void* d_out, int out_size, void* d_ws, size_t ws_size,
                              hipStream_t stream) {
    (void)in_sizes; (void)n_in; (void)out_size; (void)ws_size;
    const float* x     = (const float*)d_in[0];
    const float* W_enc = (const float*)d_in[1];
    const float* b_enc = (const float*)d_in[2];
    const float* W_dec = (const float*)d_in[3];
    const float* b_dec = (const float*)d_in[4];
    // d_in[5] = k (fixed at 64)

    float* Z  = (float*)d_ws;                                  // 4096*16384 f32
    float* tv = (float*)((char*)d_ws + (size_t)N_TOKENS * D_HIDDEN * sizeof(float));
    int*   ti = (int*)((char*)tv + (size_t)N_TOKENS * TOPK * sizeof(float));
    float* Y  = (float*)d_out;

    dim3 gA(D_HIDDEN / BN, N_TOKENS / BM);      // 128 x 32 blocks
    encode_gemm_relu<<<gA, 256, 0, stream>>>(x, W_enc, b_enc, Z);
    topk_select<<<N_TOKENS, 256, 0, stream>>>(Z, tv, ti);
    decode_sparse<<<N_TOKENS, 256, 0, stream>>>(tv, ti, W_dec, b_dec, Y);
}
